// CSMG_50629074486033
// MI455X (gfx1250) — compile-verified, matrix operand order checked by reading, not run on hardware
//
#include <hip/hip_runtime.h>
#include <hip/hip_bf16.h>
#include <math.h>

typedef __attribute__((ext_vector_type(16))) _Float16 v16h;
typedef __attribute__((ext_vector_type(8)))  float    v8f;

#if __has_builtin(__builtin_amdgcn_global_load_async_to_lds_b128) && \
    __has_builtin(__builtin_amdgcn_s_wait_asynccnt)
#define USE_ASYNC_LDS 1
typedef int vi4 __attribute__((vector_size(16)));
typedef __attribute__((address_space(1))) vi4 gvi4;   // global
typedef __attribute__((address_space(3))) vi4 lvi4;   // LDS
#endif

// ---------------------------------------------------------------------------
// Fragment gather following the CDNA5 16-bit A/B VGPR layout:
// lane&15 selects the row (A) / col (B); lane>=16 takes the k+8 half;
// element j<8 -> k = k0 + hi + j, j>=8 -> k = k0 + hi + 8 + j.
// `base` is wave-uniform (batch base only); all indexing is 32-bit element
// offsets so the backend can use SADDR+VOFFSET global addressing.
// When the k-stride is 1 the two 8-element runs are contiguous -> float4 loads.
// ---------------------------------------------------------------------------
__device__ __forceinline__ v16h load_frag(const float* __restrict__ base,
                                          unsigned off0, unsigned ks, int k0hi) {
  v16h v;
  if (ks == 1u) {
    const float4* p0 = (const float4*)(base + off0 + (unsigned)k0hi);
    const float4* p1 = (const float4*)(base + off0 + (unsigned)k0hi + 16u);
    const float4 x0 = p0[0], x1 = p0[1], x2 = p1[0], x3 = p1[1];
    v[0] = (_Float16)x0.x; v[1] = (_Float16)x0.y; v[2] = (_Float16)x0.z; v[3] = (_Float16)x0.w;
    v[4] = (_Float16)x1.x; v[5] = (_Float16)x1.y; v[6] = (_Float16)x1.z; v[7] = (_Float16)x1.w;
    v[8]  = (_Float16)x2.x; v[9]  = (_Float16)x2.y; v[10] = (_Float16)x2.z; v[11] = (_Float16)x2.w;
    v[12] = (_Float16)x3.x; v[13] = (_Float16)x3.y; v[14] = (_Float16)x3.z; v[15] = (_Float16)x3.w;
  } else {
#pragma unroll
    for (int j = 0; j < 16; ++j) {
      const unsigned k = (unsigned)(k0hi + j + ((j >= 8) ? 8 : 0));
      v[j] = (_Float16)base[off0 + k * ks];
    }
  }
  return v;
}

#define WMMA_F16(a, b, c) \
  __builtin_amdgcn_wmma_f32_16x16x32_f16(false, (a), false, (b), (short)0, (c), false, false)

// ---------------------------------------------------------------------------
// Generic strided WMMA GEMM:  C[m,n] (+= bias[m]) = sum_k A(m,k)*B(k,n)
// One wave -> 128x16 C band: 8 accumulators sharing ONE B fragment per k-step
// (8 back-to-back v_wmma_f32_16x16x32_f16; the strided B gather is amortized
// 8x, the A fragments ride the contiguous float4 path).
// Requires M%128==0, N%16==0, K%32==0 (true at every call site here).
// Element strides are 32-bit (every per-batch tensor is < 2^31 bytes).
// ---------------------------------------------------------------------------
struct GemmP {
  const float* A;
  const float* B;
  float*       C;
  const float* bias;            // per-M bias or nullptr
  unsigned a_ms, a_ks;
  unsigned b_ks, b_ns;
  unsigned c_ms, c_ns;
  long a_bs, b_bs, c_bs;        // batch strides (applied once, 64-bit)
  int  M, N, K;
  int  Mt128, Nt;               // M/128, N/16
  int  batches;
};

__global__ __launch_bounds__(256)
void gemm_f16wmma(GemmP p) {
  const int gwave = (int)((blockIdx.x * 256 + threadIdx.x) >> 5);
  const int lane  = threadIdx.x & 31;
  const int tpb   = p.Mt128 * p.Nt;
  if (gwave >= tpb * p.batches) return;
  const int b  = gwave / tpb;
  const int t  = gwave % tpb;
  const int m0 = (t / p.Nt) << 7;   // 128-row band
  const int n0 = (t % p.Nt) << 4;

  const float* A = p.A + (long)b * p.a_bs;   // wave-uniform bases
  const float* B = p.B + (long)b * p.b_bs;
  float*       C = p.C + (long)b * p.c_bs;

  const int r  = lane & 15;
  const int hi = (lane >> 4) * 8;

  const unsigned aoff0 = (unsigned)(m0 + r) * p.a_ms;
  const unsigned astep = 16u * p.a_ms;
  const unsigned boff  = (unsigned)(n0 + r) * p.b_ns;

  v8f c0 = {}, c1 = {}, c2 = {}, c3 = {}, c4 = {}, c5 = {}, c6 = {}, c7 = {};
  for (int k0 = 0; k0 < p.K; k0 += 32) {
    if (k0 + 32 < p.K) {   // gfx1250 global_prefetch path
      __builtin_prefetch(A + aoff0 + (unsigned)(k0 + 32) * p.a_ks, 0, 1);
      __builtin_prefetch(B + boff  + (unsigned)(k0 + 32) * p.b_ks, 0, 1);
    }
    const int k0hi = k0 + hi;
    const v16h bf  = load_frag(B, boff, p.b_ks, k0hi);
    v16h a;
    a = load_frag(A, aoff0,             p.a_ks, k0hi); c0 = WMMA_F16(a, bf, c0);
    a = load_frag(A, aoff0 +     astep, p.a_ks, k0hi); c1 = WMMA_F16(a, bf, c1);
    a = load_frag(A, aoff0 + 2 * astep, p.a_ks, k0hi); c2 = WMMA_F16(a, bf, c2);
    a = load_frag(A, aoff0 + 3 * astep, p.a_ks, k0hi); c3 = WMMA_F16(a, bf, c3);
    a = load_frag(A, aoff0 + 4 * astep, p.a_ks, k0hi); c4 = WMMA_F16(a, bf, c4);
    a = load_frag(A, aoff0 + 5 * astep, p.a_ks, k0hi); c5 = WMMA_F16(a, bf, c5);
    a = load_frag(A, aoff0 + 6 * astep, p.a_ks, k0hi); c6 = WMMA_F16(a, bf, c6);
    a = load_frag(A, aoff0 + 7 * astep, p.a_ks, k0hi); c7 = WMMA_F16(a, bf, c7);
  }

  const unsigned n = (unsigned)(n0 + r);
  v8f accs[8] = {c0, c1, c2, c3, c4, c5, c6, c7};
#pragma unroll
  for (int tt = 0; tt < 8; ++tt) {
#pragma unroll
    for (int v = 0; v < 8; ++v) {
      const unsigned m = (unsigned)(m0 + 16 * tt + v + hi); // vgpr v, lanes>=16 -> M=v+8
      float val = accs[tt][v];
      if (p.bias) val += p.bias[m];
      C[m * p.c_ms + n * p.c_ns] = val;
    }
  }
}

// ---------------------------------------------------------------------------
// conv3x3 (pad 1, stride 1) as implicit GEMM.
// A = weights [O, C*9] row-major (contiguous k -> float4 path),
// B = on-the-fly im2col column of 16 pixels (predicated gather, built ONCE
// per k-step and reused by 8 WMMAs -> 128 out-channels x 16 pixels per wave).
// ---------------------------------------------------------------------------
__global__ __launch_bounds__(256)
void conv3x3_wmma(const float* __restrict__ X, const float* __restrict__ Wt,
                  const float* __restrict__ bias, float* __restrict__ Y,
                  int NB, int C, int H, int W, int O) {
  const int gwave = (int)((blockIdx.x * 256 + threadIdx.x) >> 5);
  const int lane  = threadIdx.x & 31;
  const int wt    = W >> 4;
  const int ot    = O >> 7;          // 128-channel bands
  const int tpb   = H * wt * ot;
  if (gwave >= tpb * NB) return;
  int t = gwave;
  const int b  = t / tpb;        t %= tpb;
  const int hh = t / (wt * ot);  t %= (wt * ot);
  const int w0 = (t / ot) << 4;
  const int o0 = (t % ot) << 7;

  const int r  = lane & 15;
  const int hi = (lane >> 4) * 8;
  const int K  = C * 9;

  const unsigned aoff0 = (unsigned)(o0 + r) * (unsigned)K;
  const unsigned astep = 16u * (unsigned)K;
  const unsigned xplane = (unsigned)(H * W);
  const float*   Xb     = X + (long)b * C * xplane;   // wave-uniform base

  v8f c0 = {}, c1 = {}, c2 = {}, c3 = {}, c4 = {}, c5 = {}, c6 = {}, c7 = {};
  for (int k0 = 0; k0 < K; k0 += 32) {
    const int k0hi = k0 + hi;
    // im2col B fragment (predicated boundary gather), built once per k-step
    v16h bf;
#pragma unroll
    for (int j = 0; j < 16; ++j) {
      const int k  = k0hi + j + ((j >= 8) ? 8 : 0);
      const int c  = k / 9;
      const int rr = k - c * 9;
      const int dy = rr / 3;
      const int dx = rr - dy * 3;
      const int ih = hh + dy - 1;
      const int iw = w0 + r + dx - 1;
      float bv = 0.0f;
      if (ih >= 0 && ih < H && iw >= 0 && iw < W)
        bv = Xb[(unsigned)c * xplane + (unsigned)(ih * W + iw)];
      bf[j] = (_Float16)bv;
    }
    v16h a;
    a = load_frag(Wt, aoff0,             1u, k0hi); c0 = WMMA_F16(a, bf, c0);
    a = load_frag(Wt, aoff0 +     astep, 1u, k0hi); c1 = WMMA_F16(a, bf, c1);
    a = load_frag(Wt, aoff0 + 2 * astep, 1u, k0hi); c2 = WMMA_F16(a, bf, c2);
    a = load_frag(Wt, aoff0 + 3 * astep, 1u, k0hi); c3 = WMMA_F16(a, bf, c3);
    a = load_frag(Wt, aoff0 + 4 * astep, 1u, k0hi); c4 = WMMA_F16(a, bf, c4);
    a = load_frag(Wt, aoff0 + 5 * astep, 1u, k0hi); c5 = WMMA_F16(a, bf, c5);
    a = load_frag(Wt, aoff0 + 6 * astep, 1u, k0hi); c6 = WMMA_F16(a, bf, c6);
    a = load_frag(Wt, aoff0 + 7 * astep, 1u, k0hi); c7 = WMMA_F16(a, bf, c7);
  }

  float* Yb = Y + (long)b * O * xplane;   // wave-uniform base
  v8f accs[8] = {c0, c1, c2, c3, c4, c5, c6, c7};
#pragma unroll
  for (int tt = 0; tt < 8; ++tt) {
#pragma unroll
    for (int v = 0; v < 8; ++v) {
      const unsigned oc = (unsigned)(o0 + 16 * tt + v + hi);
      Yb[oc * xplane + (unsigned)(hh * W + w0 + r)] = accs[tt][v] + bias[oc];
    }
  }
}

// ---------------------------------------------------------------------------
// 2x2 max pool, stride 2
// ---------------------------------------------------------------------------
__global__ void maxpool2_k(const float* __restrict__ in, float* __restrict__ out,
                           int planes, int H, int W) {
  const int oh2 = H >> 1, ow2 = W >> 1;
  const long n = (long)planes * oh2 * ow2;
  long i = (long)blockIdx.x * blockDim.x + threadIdx.x;
  if (i >= n) return;
  const int ow = (int)(i % ow2); long t = i / ow2;
  const int oh = (int)(t % oh2);
  const long pl = t / oh2;
  const float* p = in + (pl * H + oh * 2) * (long)W + ow * 2;
  out[i] = fmaxf(fmaxf(p[0], p[1]), fmaxf(p[W], p[W + 1]));
}

// z = gamma[c]*wy*rsqrt(1+eps) + beta[c] + h   (written in place over wy)
__global__ void bn_add_k(float* __restrict__ wy, const float* __restrict__ h,
                         const float* __restrict__ gamma, const float* __restrict__ beta,
                         long n, int C, int HW) {
  long i = (long)blockIdx.x * blockDim.x + threadIdx.x;
  if (i >= n) return;
  const int c = (int)((i / HW) % C);
  const float s = rsqrtf(1.0f + 1e-5f);
  wy[i] = gamma[c] * wy[i] * s + beta[c] + h[i];
}

// ---------------------------------------------------------------------------
// in-place row softmax over rows of length L (L <= 1024, multiple of 4).
// The row is staged into LDS with gfx1250 async copies (global_load_async_
// to_lds_b128 + s_wait_asynccnt) when the toolchain exposes them, so the
// max / exp-sum / normalize passes re-read LDS instead of global.
// ---------------------------------------------------------------------------
__global__ __launch_bounds__(256)
void softmax_row_k(float* __restrict__ f, int L) {
  __shared__ float row_s[1024];
  __shared__ float red[256];
  const long row = blockIdx.x;
  const int  t   = threadIdx.x;
  float* p = f + row * (long)L;

#ifdef USE_ASYNC_LDS
  for (int j = t * 4; j < L; j += 256 * 4) {
    __builtin_amdgcn_global_load_async_to_lds_b128(
        (gvi4*)(p + j), (lvi4*)(&row_s[j]), 0, 0);
  }
  __builtin_amdgcn_s_wait_asynccnt(0);
#else
  for (int j = t; j < L; j += 256) row_s[j] = p[j];
#endif
  __syncthreads();

  float mx = -3.402823466e38f;
  for (int j = t; j < L; j += 256) mx = fmaxf(mx, row_s[j]);
  red[t] = mx; __syncthreads();
  for (int s = 128; s > 0; s >>= 1) { if (t < s) red[t] = fmaxf(red[t], red[t + s]); __syncthreads(); }
  mx = red[0]; __syncthreads();

  float sum = 0.0f;
  for (int j = t; j < L; j += 256) { float e = __expf(row_s[j] - mx); row_s[j] = e; sum += e; }
  red[t] = sum; __syncthreads();
  for (int s = 128; s > 0; s >>= 1) { if (t < s) red[t] += red[t + s]; __syncthreads(); }
  const float inv = 1.0f / red[0];
  for (int j = t; j < L; j += 256) p[j] = row_s[j] * inv;
}

// per-pixel L2 normalize over channel dim
__global__ void l2norm_ch_k(const float* __restrict__ p, float* __restrict__ xn,
                            int Bn, int C, int N) {
  const int i = blockIdx.x * blockDim.x + threadIdx.x;
  if (i >= Bn * N) return;
  const int b = i / N, n = i % N;
  const float* base = p  + (long)b * C * N + n;
  float*       o    = xn + (long)b * C * N + n;
  float s = 0.0f;
  for (int c = 0; c < C; ++c) { const float v = base[(long)c * N]; s += v * v; }
  const float inv = 1.0f / fmaxf(sqrtf(s), 1e-12f);
  for (int c = 0; c < C; ++c) o[(long)c * N] = base[(long)c * N] * inv;
}

// normalize centroid rows
__global__ void cnorm_k(const float* __restrict__ cent, float* __restrict__ cn,
                        int K, int D) {
  const int k = blockIdx.x * blockDim.x + threadIdx.x;
  if (k >= K) return;
  float s = 0.0f;
  for (int d = 0; d < D; ++d) { const float v = cent[k * D + d]; s += v * v; }
  const float inv = 1.0f / fmaxf(sqrtf(s), 1e-12f);
  for (int d = 0; d < D; ++d) cn[k * D + d] = cent[k * D + d] * inv;
}

// per-pixel: logits->softmax over 4 clusters, sim=relu(cn.xn), wgt=soft*sim
__global__ void assign_k(const float* __restrict__ xn, const float* __restrict__ node_w,
                         const float* __restrict__ node_b, const float* __restrict__ cn,
                         float* __restrict__ sim_out, float* __restrict__ wgt,
                         int Bn, int D, int N) {
  const int i = blockIdx.x * blockDim.x + threadIdx.x;
  if (i >= Bn * N) return;
  const int b = i / N, n = i % N;
  const float* xb = xn + (long)b * D * N + n;
  float logit[4], sim[4];
#pragma unroll
  for (int k = 0; k < 4; ++k) {
    float lg = node_b[k], sm = 0.0f;
    for (int d = 0; d < D; ++d) {
      const float xv = xb[(long)d * N];
      lg += node_w[k * D + d] * xv;
      sm += cn[k * D + d] * xv;
    }
    logit[k] = lg; sim[k] = fmaxf(sm, 0.0f);
  }
  float mx = fmaxf(fmaxf(logit[0], logit[1]), fmaxf(logit[2], logit[3]));
  float den = 0.0f;
#pragma unroll
  for (int k = 0; k < 4; ++k) { logit[k] = __expf(logit[k] - mx); den += logit[k]; }
  const float inv = 1.0f / den;
#pragma unroll
  for (int k = 0; k < 4; ++k) {
    const long o = ((long)b * 4 + k) * N + n;
    sim_out[o] = sim[k];
    wgt[o]     = logit[k] * inv * sim[k];
  }
}

// d[b,k,dd] = sum_n xn[b,dd,n] * wgt[b,k,n]
__global__ void vlad_d_k(const float* __restrict__ xn, const float* __restrict__ wgt,
                         float* __restrict__ d, int Bn, int D, int N) {
  const int i = blockIdx.x * blockDim.x + threadIdx.x;
  if (i >= Bn * 4 * D) return;
  const int dd = i % D; int t = i / D;
  const int k = t % 4;  const int b = t / 4;
  const float* xb = xn  + ((long)b * D + dd) * N;
  const float* wb = wgt + ((long)b * 4 + k) * N;
  float s = 0.0f;
  for (int n = 0; n < N; ++n) s += xb[n] * wb[n];
  d[i] = s;
}

// L2-normalize rows of length 256 (one block per row)
__global__ void dnorm_k(const float* __restrict__ din, float* __restrict__ dout, int D) {
  __shared__ float red[256];
  const int row = blockIdx.x, t = threadIdx.x;
  const float v = din[(long)row * D + t];
  red[t] = v * v; __syncthreads();
  for (int s = 128; s > 0; s >>= 1) { if (t < s) red[t] += red[t + s]; __syncthreads(); }
  const float inv = 1.0f / fmaxf(sqrtf(red[0]), 1e-12f);
  dout[(long)row * D + t] = v * inv;
}

// flatten + L2 normalize per batch (L = 1024)
__global__ void dflat_k(const float* __restrict__ dn, float* __restrict__ out, int L) {
  __shared__ float red[256];
  const int b = blockIdx.x, t = threadIdx.x;
  float s = 0.0f;
  for (int j = t; j < L; j += 256) { const float v = dn[(long)b * L + j]; s += v * v; }
  red[t] = s; __syncthreads();
  for (int st = 128; st > 0; st >>= 1) { if (t < st) red[t] += red[t + st]; __syncthreads(); }
  const float inv = 1.0f / fmaxf(sqrtf(red[0]), 1e-12f);
  for (int j = t; j < L; j += 256) out[(long)b * L + j] = dn[(long)b * L + j] * inv;
}

// ---------------------------------------------------------------------------
extern "C" void kernel_launch(void* const* d_in, const int* in_sizes, int n_in,
                              void* d_out, int out_size, void* d_ws, size_t ws_size,
                              hipStream_t stream) {
  (void)in_sizes; (void)n_in; (void)out_size; (void)ws_size;
  const float* x        = (const float*)d_in[0];
  const float* conv1_w  = (const float*)d_in[1];
  const float* conv1_b  = (const float*)d_in[2];
  const float* theta_w  = (const float*)d_in[3];
  const float* theta_b  = (const float*)d_in[4];
  const float* phi_w    = (const float*)d_in[5];
  const float* phi_b    = (const float*)d_in[6];
  const float* g_w      = (const float*)d_in[7];
  const float* g_b      = (const float*)d_in[8];
  const float* wz_w     = (const float*)d_in[9];
  const float* wz_b     = (const float*)d_in[10];
  const float* bn_gamma = (const float*)d_in[11];
  const float* bn_beta  = (const float*)d_in[12];
  const float* cent     = (const float*)d_in[13];
  const float* node_w   = (const float*)d_in[14];
  const float* node_b   = (const float*)d_in[15];
  float* out = (float*)d_out;

  const int B = 16, C = 256, H = 64, W = 64, I = 128;
  const long HW = (long)H * W;          // 4096
  const long N2 = HW / 4;               // 1024 (pooled)

  float* ws = (float*)d_ws;
  long o = 0;
  float* h    = ws + o; o += (long)B * C * HW;
  float* th   = ws + o; o += (long)B * I * HW;
  float* phif = ws + o; o += (long)B * I * HW;
  float* gf   = ws + o; o += (long)B * I * HW;
  float* ph   = ws + o; o += (long)B * I * N2;
  float* gx   = ws + o; o += (long)B * I * N2;
  float* f    = ws + o; o += (long)B * HW * N2;
  float* y    = ws + o; o += (long)B * I * HW;
  float* wy   = ws + o; o += (long)B * C * HW;   // becomes z in place
  float* p    = ws + o; o += (long)B * C * N2;
  float* xn   = ws + o; o += (long)B * C * N2;
  float* wgt  = ws + o; o += (long)B * 4 * N2;
  float* cn   = ws + o; o += 4 * C;
  float* dtmp = ws + o; o += (long)B * 4 * C;

  auto gemm_blocks = [](long waves) { return (unsigned)((waves + 7) / 8); };

  // 1) conv3x3 -> h   (waves: B * H * (W/16) * (O/128))
  {
    long waves = (long)B * H * (W / 16) * (C / 128);
    conv3x3_wmma<<<gemm_blocks(waves), 256, 0, stream>>>(x, conv1_w, conv1_b, h, B, C, H, W, C);
  }

  // 2-4) theta/phi/g 1x1 convs
  {
    GemmP g{};
    g.a_ms = (unsigned)C; g.a_ks = 1; g.b_ks = (unsigned)HW; g.b_ns = 1;
    g.c_ms = (unsigned)HW; g.c_ns = 1;
    g.a_bs = 0; g.b_bs = (long)C * HW; g.c_bs = (long)I * HW;
    g.M = I; g.N = (int)HW; g.K = C; g.Mt128 = I / 128; g.Nt = (int)(HW / 16); g.batches = B;
    long waves = (long)g.Mt128 * g.Nt * B;
    g.A = theta_w; g.B = h; g.C = th;   g.bias = theta_b;
    gemm_f16wmma<<<gemm_blocks(waves), 256, 0, stream>>>(g);
    g.A = phi_w;   g.C = phif; g.bias = phi_b;
    gemm_f16wmma<<<gemm_blocks(waves), 256, 0, stream>>>(g);
    g.A = g_w;     g.C = gf;   g.bias = g_b;
    gemm_f16wmma<<<gemm_blocks(waves), 256, 0, stream>>>(g);
  }

  // 5-6) maxpool phi/g to 32x32
  {
    long n = (long)B * I * N2;
    maxpool2_k<<<(unsigned)((n + 255) / 256), 256, 0, stream>>>(phif, ph, B * I, H, W);
    maxpool2_k<<<(unsigned)((n + 255) / 256), 256, 0, stream>>>(gf,   gx, B * I, H, W);
  }

  // 7) f[b,n,m] = sum_i th[b,i,n]*ph[b,i,m]   (M=4096, N=1024, K=128)
  {
    GemmP g{};
    g.A = th; g.B = ph; g.C = f; g.bias = nullptr;
    g.a_ms = 1; g.a_ks = (unsigned)HW; g.b_ks = (unsigned)N2; g.b_ns = 1;
    g.c_ms = (unsigned)N2; g.c_ns = 1;
    g.a_bs = (long)I * HW; g.b_bs = (long)I * N2; g.c_bs = HW * N2;
    g.M = (int)HW; g.N = (int)N2; g.K = I; g.Mt128 = (int)(HW / 128); g.Nt = (int)(N2 / 16); g.batches = B;
    gemm_f16wmma<<<gemm_blocks((long)g.Mt128 * g.Nt * B), 256, 0, stream>>>(g);
  }

  // 8) softmax over last axis of f (async-LDS staged)
  softmax_row_k<<<(unsigned)(B * HW), 256, 0, stream>>>(f, (int)N2);

  // 9) y[b,i,n] = sum_m f[b,n,m]*gx[b,i,m]   (M=128, N=4096, K=1024; b_ks=1 -> float4 path)
  {
    GemmP g{};
    g.A = gx; g.B = f; g.C = y; g.bias = nullptr;
    g.a_ms = (unsigned)N2; g.a_ks = 1; g.b_ks = 1; g.b_ns = (unsigned)N2;
    g.c_ms = (unsigned)HW; g.c_ns = 1;
    g.a_bs = (long)I * N2; g.b_bs = HW * N2; g.c_bs = (long)I * HW;
    g.M = I; g.N = (int)HW; g.K = (int)N2; g.Mt128 = I / 128; g.Nt = (int)(HW / 16); g.batches = B;
    gemm_f16wmma<<<gemm_blocks((long)g.Mt128 * g.Nt * B), 256, 0, stream>>>(g);
  }

  // 10) wy[b,o,hw] = sum_i wz_w[o,i]*y[b,i,hw] + wz_b[o]
  {
    GemmP g{};
    g.A = wz_w; g.B = y; g.C = wy; g.bias = wz_b;
    g.a_ms = (unsigned)I; g.a_ks = 1; g.b_ks = (unsigned)HW; g.b_ns = 1;
    g.c_ms = (unsigned)HW; g.c_ns = 1;
    g.a_bs = 0; g.b_bs = (long)I * HW; g.c_bs = (long)C * HW;
    g.M = C; g.N = (int)HW; g.K = I; g.Mt128 = C / 128; g.Nt = (int)(HW / 16); g.batches = B;
    gemm_f16wmma<<<gemm_blocks((long)g.Mt128 * g.Nt * B), 256, 0, stream>>>(g);
  }

  // 11) z = BN(wy) + h  (in place over wy)
  {
    long n = (long)B * C * HW;
    bn_add_k<<<(unsigned)((n + 255) / 256), 256, 0, stream>>>(wy, h, bn_gamma, bn_beta, n, C, (int)HW);
  }

  // 12) p = maxpool2(z)
  {
    long n = (long)B * C * N2;
    maxpool2_k<<<(unsigned)((n + 255) / 256), 256, 0, stream>>>(wy, p, B * C, H, W);
  }

  // 13) xn = l2norm over channels
  l2norm_ch_k<<<(unsigned)((B * N2 + 255) / 256), 256, 0, stream>>>(p, xn, B, C, (int)N2);

  // 14) normalized centroids
  cnorm_k<<<1, 64, 0, stream>>>(cent, cn, 4, C);

  // 15) sim_scores (-> out) + wgt
  assign_k<<<(unsigned)((B * N2 + 255) / 256), 256, 0, stream>>>(
      xn, node_w, node_b, cn, out, wgt, B, C, (int)N2);

  // 16) d accumulation
  {
    long n = (long)B * 4 * C;
    vlad_d_k<<<(unsigned)((n + 255) / 256), 256, 0, stream>>>(xn, wgt, dtmp, B, C, (int)N2);
  }

  // 17) d row-normalize -> out[65536..]
  float* out_d = out + (long)B * 4 * N2;
  dnorm_k<<<B * 4, 256, 0, stream>>>(dtmp, out_d, C);

  // 18) flatten-normalize -> out[65536+16384..]
  float* out_fl = out_d + (long)B * 4 * C;
  dflat_k<<<B, 256, 0, stream>>>(out_d, out_fl, 4 * C);
}
